// DeepRNN_5686536700312
// MI455X (gfx1250) — matrix operand order repaired and checked by reading, not compiled, not run on hardware
//
#include <hip/hip_runtime.h>

// ---------------------------------------------------------------------------
// DeepRNN on MI455X (gfx1250): persistent bf16-WMMA kernel.
//   B=64, T=512, I=512, H=1024, L=4, O=512
//   per (t,layer): pre = A1 @ W1^T + A2 @ W2^T + b ; h = tanh(pre)
//   515 GFLOP total, 2048 strictly sequential GEMM steps -> latency bound.
//   Weights (15 MB bf16) live in L2; sync via in-kernel global barrier.
// ---------------------------------------------------------------------------

typedef __attribute__((ext_vector_type(16))) __bf16 v16bf;
typedef __attribute__((ext_vector_type(8)))  __bf16 v8bf;
typedef __attribute__((ext_vector_type(8)))  float  v8f;

#define B_    64
#define T_    512
#define I_    512
#define H_    1024
#define L_    4
#define O_    512
#define NBLK  64      // persistent workgroups; each owns a 16-wide N strip
#define NWAVE 8       // wave32 waves per workgroup (K-split)

// ---------------------------------------------------------------------------
// Prologue kernels
// ---------------------------------------------------------------------------
__global__ void f32_to_bf16_kernel(const float* __restrict__ src,
                                   __bf16* __restrict__ dst, size_t n) {
  size_t stride = (size_t)gridDim.x * blockDim.x;
  for (size_t i = (size_t)blockIdx.x * blockDim.x + threadIdx.x; i < n; i += stride)
    dst[i] = (__bf16)src[i];
}

// W: [nout][kin] row-major  ->  Wt: [kin][nout] row-major (bf16)
__global__ void transpose_to_bf16_kernel(const float* __restrict__ W,
                                         __bf16* __restrict__ Wt,
                                         int nout, int kin) {
  size_t n = (size_t)nout * kin;
  size_t stride = (size_t)gridDim.x * blockDim.x;
  for (size_t i = (size_t)blockIdx.x * blockDim.x + threadIdx.x; i < n; i += stride) {
    size_t k = i / (size_t)nout;
    size_t j = i % (size_t)nout;
    Wt[i] = (__bf16)W[j * (size_t)kin + k];
  }
}

__global__ void sync_reset_kernel(unsigned* sync) {
  if (threadIdx.x == 0 && blockIdx.x == 0) { sync[0] = 0u; sync[1] = 0u; }
}

// ---------------------------------------------------------------------------
// Device helpers
// ---------------------------------------------------------------------------

// Grid-wide arrive/release barrier (device scope).
__device__ __forceinline__ void grid_sync(unsigned* bar, unsigned* gen, unsigned nblocks) {
  __threadfence();
  __syncthreads();
  if (threadIdx.x == 0) {
    unsigned g = __hip_atomic_load(gen, __ATOMIC_RELAXED, __HIP_MEMORY_SCOPE_AGENT);
    unsigned a = __hip_atomic_fetch_add(bar, 1u, __ATOMIC_ACQ_REL, __HIP_MEMORY_SCOPE_AGENT);
    if (a == nblocks - 1u) {
      __hip_atomic_store(bar, 0u, __ATOMIC_RELAXED, __HIP_MEMORY_SCOPE_AGENT);
      __hip_atomic_store(gen, g + 1u, __ATOMIC_RELEASE, __HIP_MEMORY_SCOPE_AGENT);
    } else {
      while (__hip_atomic_load(gen, __ATOMIC_ACQUIRE, __HIP_MEMORY_SCOPE_AGENT) == g)
        __builtin_amdgcn_s_sleep(2);
    }
  }
  __syncthreads();
  __threadfence();
}

// B fragment (K=32 x N=16, bf16): lane<16 holds column j0+L, K {0..7, 16..23};
// lane>=16 holds column j0+L-16, K {8..15, 24..31}. Strided scalar loads.
__device__ __forceinline__ v16bf load_b_frag(const __bf16* __restrict__ Wt,
                                             int k0, int ncols, int j0, int lane) {
  v16bf r;
#pragma unroll
  for (int s = 0; s < 16; ++s) {
    int kk = (s & 7) + ((s & 8) ? 16 : 0) + ((lane & 16) ? 8 : 0);
    r[s] = Wt[(size_t)(k0 + kk) * (size_t)ncols + j0 + (lane & 15)];
  }
  return r;
}

// A fragment (M=16 x K=32, bf16): lane<16 holds row m16+L, K {0..7, 16..23};
// lane>=16 holds row m16+L-16, K {8..15, 24..31}. Two 16-byte loads.
__device__ __forceinline__ v16bf load_a_frag(const __bf16* __restrict__ A,
                                             size_t rstride, int m, int k0, int lane) {
  int row = m * 16 + (lane & 15);
  const __bf16* p = A + (size_t)row * rstride + (size_t)(k0 + ((lane & 16) ? 8 : 0));
  v8bf lo = *(const v8bf*)p;
  v8bf hi = *(const v8bf*)(p + 16);
  v16bf r;
#pragma unroll
  for (int i = 0; i < 8; ++i) { r[i] = lo[i]; r[i + 8] = hi[i]; }
  return r;
}

// One K-segment of the fused GEMM: this wave takes K-chunks kc = wave, wave+8, ...
// acc[m] accumulates the (16m..16m+15) x (j0..j0+15) f32 tile.
__device__ __forceinline__ void gemm_seg(const __bf16* __restrict__ A, size_t rstride,
                                         const __bf16* __restrict__ Wt, int ncols, int Ks,
                                         int j0, int wave, int lane, v8f* acc) {
  int nchunk = Ks >> 5;
  for (int kc = wave; kc < nchunk; kc += NWAVE) {
    int k0 = kc << 5;
    v16bf bf = load_b_frag(Wt, k0, ncols, j0, lane);
    // prefetch this wave's next weight chunk (-> global_prefetch_b8)
    __builtin_prefetch(Wt + (size_t)(k0 + NWAVE * 32 + lane) * (size_t)ncols + j0, 0, 1);
#pragma unroll
    for (int m = 0; m < 4; ++m) {
      v16bf af = load_a_frag(A, rstride, m, k0, lane);
      acc[m] = __builtin_amdgcn_wmma_f32_16x16x32_bf16(
          false, af, false, bf, (short)0, acc[m], false, false);
    }
  }
}

// Cross-wave K-split reduction through LDS, then bias (+tanh) epilogue.
// dstH: bf16 hidden store (stride ncols)  |  dstF: f32 output store.
__device__ __forceinline__ void reduce_store(float* red, v8f* acc, int wave, int lane,
                                             const float* bias1, const float* bias2,
                                             __bf16* dstH, float* dstF,
                                             int ncols, int j0, bool do_tanh) {
#pragma unroll
  for (int m = 0; m < 4; ++m)
#pragma unroll
    for (int r = 0; r < 8; ++r)
      red[wave * 1024 + m * 256 + r * 32 + lane] = acc[m][r];
  __syncthreads();
  if (wave < 4) {
    int m = wave;
    int j = j0 + (lane & 15);
    float bias = (bias1 ? bias1[j] : 0.0f) + (bias2 ? bias2[j] : 0.0f);
#pragma unroll
    for (int r = 0; r < 8; ++r) {
      float s = bias;
#pragma unroll
      for (int wv = 0; wv < NWAVE; ++wv)
        s += red[wv * 1024 + m * 256 + r * 32 + lane];
      // C/D layout: VGPR r, lane<16 -> (M=r, N=lane); lane>=16 -> (M=8+r, N=lane-16)
      int row = m * 16 + r + ((lane & 16) ? 8 : 0);
      if (do_tanh) {
        dstH[(size_t)row * (size_t)ncols + j] = (__bf16)tanhf(s);
      } else {
        dstF[(size_t)row * (size_t)ncols + j] = s;
      }
    }
  }
}

// ---------------------------------------------------------------------------
// Persistent RNN kernel: 64 WGs x 256 threads. Each WG owns 16 output columns.
// Each (t, layer) step: fused 2-segment GEMM, LDS K-split reduce, tanh, store,
// grid barrier. Final FC done by WGs 0..31.
// ---------------------------------------------------------------------------
__global__ __launch_bounds__(256, 1)
void rnn_persistent_kernel(const __bf16* __restrict__ xb,      // [B][T][I]
                           const __bf16* __restrict__ Wih0t,   // [I][H]
                           const __bf16* __restrict__ Wiht,    // [3][H][H] (K-major)
                           const __bf16* __restrict__ Whht,    // [4][H][H] (K-major)
                           const __bf16* __restrict__ Woutt,   // [H][O]
                           const float* __restrict__ b_ih0,
                           const float* __restrict__ b_ih,     // [3][H]
                           const float* __restrict__ b_hh,     // [4][H]
                           const float* __restrict__ b_out,    // [O]
                           __bf16* __restrict__ hbuf,          // [2][L][B][H]
                           float* __restrict__ out,            // [B][O]
                           unsigned* bar, unsigned* gen) {
  __shared__ float red[NWAVE * 4 * 256];   // 32 KB of the 320 KB WGP LDS
  const int wave = threadIdx.x >> 5;
  const int lane = threadIdx.x & 31;
  const int j0 = blockIdx.x * 16;
  const size_t HB4 = (size_t)L_ * B_ * H_;

  for (int t = 0; t < T_; ++t) {
    __bf16* hprev = hbuf + (size_t)(t & 1) * HB4;        // h(t-1)
    __bf16* hcur  = hbuf + (size_t)((t + 1) & 1) * HB4;  // h(t)
    for (int c = 0; c < L_; ++c) {
      v8f acc[4] = {};
      if (c == 0) {
        // segment 1: x_t @ W_ih0^T   (A row stride T*I, base offset t*I)
        gemm_seg(xb + (size_t)t * I_, (size_t)T_ * I_, Wih0t, H_, I_, j0, wave, lane, acc);
      } else {
        // segment 1: h_t[c-1] @ W_ih[c-1]^T
        gemm_seg(hcur + (size_t)(c - 1) * B_ * H_, (size_t)H_,
                 Wiht + (size_t)(c - 1) * H_ * H_, H_, H_, j0, wave, lane, acc);
      }
      // segment 2: h_{t-1}[c] @ W_hh[c]^T
      gemm_seg(hprev + (size_t)c * B_ * H_, (size_t)H_,
               Whht + (size_t)c * H_ * H_, H_, H_, j0, wave, lane, acc);

      const float* bi = (c == 0) ? b_ih0 : (b_ih + (size_t)(c - 1) * H_);
      reduce_store(red, acc, wave, lane, bi, b_hh + (size_t)c * H_,
                   hcur + (size_t)c * B_ * H_, nullptr, H_, j0, true);
      grid_sync(bar, gen, NBLK);
    }
  }

  // Final FC: out = h_T[L-1] @ W_out^T + b_out  (N=512 -> WGs 0..31)
  if (blockIdx.x < O_ / 16) {
    v8f acc[4] = {};
    const __bf16* hfin = hbuf + (size_t)(T_ & 1) * HB4 + (size_t)(L_ - 1) * B_ * H_;
    gemm_seg(hfin, (size_t)H_, Woutt, O_, H_, j0, wave, lane, acc);
    reduce_store(red, acc, wave, lane, b_out, nullptr, nullptr, out, O_, j0, false);
  }
}

// ---------------------------------------------------------------------------
// Host launcher
// ---------------------------------------------------------------------------
extern "C" void kernel_launch(void* const* d_in, const int* in_sizes, int n_in,
                              void* d_out, int out_size, void* d_ws, size_t ws_size,
                              hipStream_t stream) {
  const float* x     = (const float*)d_in[0];
  const float* h0    = (const float*)d_in[1];
  const float* W_ih0 = (const float*)d_in[2];
  const float* b_ih0 = (const float*)d_in[3];
  const float* W_ih  = (const float*)d_in[4];
  const float* b_ih  = (const float*)d_in[5];
  const float* W_hh  = (const float*)d_in[6];
  const float* b_hh  = (const float*)d_in[7];
  const float* W_out = (const float*)d_in[8];
  const float* b_out = (const float*)d_in[9];
  (void)in_sizes; (void)n_in; (void)out_size; (void)ws_size;

  // Workspace carve-up (all sizes power-of-two-ish; base is hipMalloc aligned)
  char* p = (char*)d_ws;
  __bf16* xb    = (__bf16*)p; p += (size_t)B_ * T_ * I_ * 2;      // 33.5 MB
  __bf16* Wih0t = (__bf16*)p; p += (size_t)I_ * H_ * 2;           //  1 MB
  __bf16* Wiht  = (__bf16*)p; p += (size_t)3 * H_ * H_ * 2;       //  6 MB
  __bf16* Whht  = (__bf16*)p; p += (size_t)4 * H_ * H_ * 2;       //  8 MB
  __bf16* Woutt = (__bf16*)p; p += (size_t)H_ * O_ * 2;           //  1 MB
  __bf16* hbuf  = (__bf16*)p; p += (size_t)2 * L_ * B_ * H_ * 2;  //  1 MB
  unsigned* sync = (unsigned*)p;

  // Prologue: bf16 conversions / weight transposes / barrier reset
  f32_to_bf16_kernel<<<2048, 256, 0, stream>>>(x, xb, (size_t)B_ * T_ * I_);
  f32_to_bf16_kernel<<<512, 256, 0, stream>>>(h0, hbuf, (size_t)L_ * B_ * H_); // buffer 0 = h(t=-1)
  transpose_to_bf16_kernel<<<512, 256, 0, stream>>>(W_ih0, Wih0t, H_, I_);
  for (int c = 0; c < 3; ++c)
    transpose_to_bf16_kernel<<<512, 256, 0, stream>>>(
        W_ih + (size_t)c * H_ * H_, Wiht + (size_t)c * H_ * H_, H_, H_);
  for (int c = 0; c < 4; ++c)
    transpose_to_bf16_kernel<<<512, 256, 0, stream>>>(
        W_hh + (size_t)c * H_ * H_, Whht + (size_t)c * H_ * H_, H_, H_);
  transpose_to_bf16_kernel<<<512, 256, 0, stream>>>(W_out, Woutt, O_, H_);
  sync_reset_kernel<<<1, 32, 0, stream>>>(sync);

  // Persistent RNN
  rnn_persistent_kernel<<<NBLK, 256, 0, stream>>>(
      xb, Wih0t, Wiht, Whht, Woutt, b_ih0, b_ih, b_hh, b_out,
      hbuf, (float*)d_out, sync, sync + 1);
}
